// HybridSTPAttention_57732950393320
// MI455X (gfx1250) — matrix operand order repaired and checked by reading, not compiled, not run on hardware
//
#include <hip/hip_runtime.h>
#include <math.h>

typedef __attribute__((ext_vector_type(16))) _Float16 v16h;
typedef __attribute__((ext_vector_type(8)))  _Float16 v8h;
typedef __attribute__((ext_vector_type(8)))  float    v8f;
typedef __attribute__((ext_vector_type(4)))  float    v4f;

#define Lq 2048
#define Dm 1024
#define Hh 16
#define dh 64
#define SCALE 0.125f   // 1/sqrt(64)

// ---------------- WMMA fragment loads (all b128 from LDS) ----------------
// A 16x32 f16 (ISA 7.12.2): lane<16 row=lane K={0..7,16..23}; lane>=16 K shifts +8.
static __device__ __forceinline__ v16h load_a_frag(const _Float16* lds, int stride,
                                                   int row, int k0, int lane) {
  const _Float16* base = lds + row * stride + k0 + ((lane & 16) ? 8 : 0);
  v8h a0 = *(const v8h*)(base);
  v8h a1 = *(const v8h*)(base + 16);
  return __builtin_shufflevector(a0, a1, 0,1,2,3,4,5,6,7,8,9,10,11,12,13,14,15);
}
// B 32x16 f16: lane<16 col=lane K={0..7,16..23}; lane>=16 K shifts +8.  LDS layout [col][K].
static __device__ __forceinline__ v16h load_b_frag(const _Float16* lds, int stride,
                                                   int col, int k0, int lane) {
  const _Float16* base = lds + col * stride + k0 + ((lane & 16) ? 8 : 0);
  v8h b0 = *(const v8h*)(base);
  v8h b1 = *(const v8h*)(base + 16);
  return __builtin_shufflevector(b0, b1, 0,1,2,3,4,5,6,7,8,9,10,11,12,13,14,15);
}

static __device__ __forceinline__ v8h pack8(const float* s) {
  v4f f0 = *(const v4f*)(s);
  v4f f1 = *(const v4f*)(s + 4);
  v8h h;
#pragma unroll
  for (int i = 0; i < 4; ++i) { h[i] = (_Float16)f0[i]; h[4 + i] = (_Float16)f1[i]; }
  return h;
}

// ---------------- WMMA GEMM: C = A[MxK] * W[KxN] + bias ----------------
// Requires M%64==0, N%128==0, K%32==0.  mode 0: bias  mode 1: bias+GELU(exact)
__global__ __launch_bounds__(256)
void gemm_wmma(const float* __restrict__ A, const float* __restrict__ W,
               const float* __restrict__ bias, float* __restrict__ C,
               int M, int N, int K, int mode) {
  __shared__ __align__(16) _Float16 At[64 * 40];    // [row][K]
  __shared__ __align__(16) _Float16 Bt[128 * 40];   // [col][K]
  const int SA = 40, SB = 40;
  int tid = threadIdx.x, lane = tid & 31, wid = tid >> 5;
  int wm = wid & 1, wn = wid >> 1;                  // 2x4 waves, 32x32 each
  int bm = blockIdx.x * 64, bn = blockIdx.y * 128;

  // A stage: thread reads 8 contiguous floats of one A row
  int arow = tid >> 2, akc = (tid & 3) * 8;
  const float* aPtr = A + (size_t)(bm + arow) * K + akc;
  _Float16*    aDst = At + arow * SA + akc;
  // B stage: thread reads 16 contiguous floats of one W row, scatters to Bt[col][K]
  int bkr = tid >> 3, bcc = (tid & 7) * 16;
  const float* wPtr = W + (size_t)bkr * N + bn + bcc;
  _Float16*    bDst = Bt + bcc * SB + bkr;          // + e*SB for col bcc+e
  const size_t wStep = (size_t)32 * N;

  v8f acc[2][2] = {};

  for (int k0 = 0; k0 < K; k0 += 32) {
    *(v8h*)aDst = pack8(aPtr);                      // 2x b128 load + packed store
    {
      v4f f[4];
#pragma unroll
      for (int m = 0; m < 4; ++m) f[m] = *(const v4f*)(wPtr + m * 4);
#pragma unroll
      for (int m = 0; m < 4; ++m)
#pragma unroll
        for (int e = 0; e < 4; ++e)
          bDst[(m * 4 + e) * SB] = (_Float16)f[m][e];  // imm-offset ds_store_b16
    }
    if (k0 + 32 < K) {
      __builtin_prefetch(aPtr + 32, 0, 1);
      __builtin_prefetch(wPtr + wStep, 0, 1);
    }
    __syncthreads();

    v16h af[2], bf[2];
#pragma unroll
    for (int mt = 0; mt < 2; ++mt)
      af[mt] = load_a_frag(At, SA, wm * 32 + mt * 16 + (lane & 15), 0, lane);
#pragma unroll
    for (int nt = 0; nt < 2; ++nt)
      bf[nt] = load_b_frag(Bt, SB, wn * 32 + nt * 16 + (lane & 15), 0, lane);
#pragma unroll
    for (int mt = 0; mt < 2; ++mt)
#pragma unroll
      for (int nt = 0; nt < 2; ++nt)
        acc[mt][nt] = __builtin_amdgcn_wmma_f32_16x16x32_f16(
            false, af[mt], false, bf[nt], (short)0, acc[mt][nt], false, false);
    __syncthreads();
    aPtr += 32;
    wPtr += wStep;
  }

  int hi8 = (lane & 16) ? 8 : 0;
#pragma unroll
  for (int mt = 0; mt < 2; ++mt)
#pragma unroll
    for (int nt = 0; nt < 2; ++nt) {
      int col = bn + wn * 32 + nt * 16 + (lane & 15);
      float b = bias[col];
#pragma unroll
      for (int r = 0; r < 8; ++r) {
        int row = bm + wm * 32 + mt * 16 + r + hi8;
        float v = acc[mt][nt][r] + b;
        if (mode == 1) v = 0.5f * v * (1.0f + erff(v * 0.70710678118f));
        C[(size_t)row * N + col] = v;
      }
    }
}

// ---------------- repack qkv (L,3D) -> q,k,v each (H,L,d) ----------------
__global__ void repack_qkv(const float* __restrict__ qkv, float* __restrict__ q,
                           float* __restrict__ k, float* __restrict__ v) {
  int idx = blockIdx.x * blockDim.x + threadIdx.x;
  if (idx >= Lq * 3 * Dm) return;
  int l = idx / (3 * Dm);
  int c = idx - l * 3 * Dm;
  int s = c / Dm;
  int hc = c - s * Dm;
  int h = hc >> 6, i = hc & 63;
  float val = qkv[idx];
  float* dst = (s == 0) ? q : (s == 1) ? k : v;
  dst[(((size_t)h * Lq + l) << 6) + i] = val;
}

// ---------------- causal flash attention, one WG per (qblock, head) ----------------
__global__ __launch_bounds__(128)
void flash_attn(const float* __restrict__ q, const float* __restrict__ k,
                const float* __restrict__ v, float* __restrict__ y) {
  __shared__ __align__(16) _Float16 Qh[64 * 72];   // [row][d]  (A for S)
  __shared__ __align__(16) _Float16 Kc[64 * 72];   // [j][d]    (B for S)
  __shared__ __align__(16) _Float16 Vc[64 * 72];   // [i][j]    (B for PV)
  __shared__ __align__(16) _Float16 Pl[64 * 72];   // [row][j]  (A for PV)
  __shared__ __align__(16) float    Sls[64 * 68];  // scores f32
  __shared__ float mrow[64], lrow[64], crow[64];
  const int ST = 72, SS = 68;
  int tid = threadIdx.x, lane = tid & 31, w = tid >> 5;
  int qb = blockIdx.x, h = blockIdx.y;
  const float* qg = q + ((size_t)h * Lq + qb * 64) * dh;

  // stage Q tile (row copy, coalesced + packed)
  {
    int r = tid >> 1, c0 = (tid & 1) * 32;
#pragma unroll
    for (int m = 0; m < 4; ++m)
      *(v8h*)(Qh + r * ST + c0 + m * 8) = pack8(qg + (size_t)r * dh + c0 + m * 8);
  }
  if (tid < 64) { mrow[tid] = -3.0e38f; lrow[tid] = 0.0f; }
  v8f Oa[4] = {};
  __syncthreads();

  int arow = w * 16 + (lane & 15);
  int hi8 = (lane & 16) ? 8 : 0;

  // hoisted stage pointers, advanced by 64 rows per jb
  int sr = tid >> 1, sc0 = (tid & 1) * 32;          // K stage: row copy
  const float* kPtr = k + ((size_t)h * Lq + sr) * dh + sc0;
  int vj = tid >> 1, vi0 = (tid & 1) * 32;          // V stage: LDS-side transpose
  const float* vPtr = v + ((size_t)h * Lq + vj) * dh + vi0;
  _Float16* vDst = Vc + vi0 * ST + vj;              // + e*ST for col vi0+e
  const size_t kvStep = (size_t)64 * dh;

  for (int jb = 0; jb <= qb; ++jb) {
    // stage K rows directly (Kc[j][d] is the B layout for S)
#pragma unroll
    for (int m = 0; m < 4; ++m)
      *(v8h*)(Kc + sr * ST + sc0 + m * 8) = pack8(kPtr + m * 8);
    // stage V transposed via LDS scatter: contiguous global reads, imm-offset b16 stores
    {
      v4f f[8];
#pragma unroll
      for (int m = 0; m < 8; ++m) f[m] = *(const v4f*)(vPtr + m * 4);
#pragma unroll
      for (int m = 0; m < 8; ++m)
#pragma unroll
        for (int e = 0; e < 4; ++e)
          vDst[(m * 4 + e) * ST] = (_Float16)f[m][e];
    }
    __syncthreads();

    // S strip = Q K^T (scaled, causal)
#pragma unroll
    for (int nt = 0; nt < 4; ++nt) {
      v8f s = {};
#pragma unroll
      for (int k0 = 0; k0 < 64; k0 += 32) {
        v16h a = load_a_frag(Qh, ST, arow, k0, lane);
        v16h b = load_b_frag(Kc, ST, nt * 16 + (lane & 15), k0, lane);
        s = __builtin_amdgcn_wmma_f32_16x16x32_f16(false, a, false, b, (short)0, s, false, false);
      }
#pragma unroll
      for (int r = 0; r < 8; ++r) {
        int rl = w * 16 + r + hi8;
        int cl = nt * 16 + (lane & 15);
        float sv = s[r] * SCALE;
        if (jb * 64 + cl > qb * 64 + rl) sv = -3.0e38f;   // exp underflow -> exact causal
        Sls[rl * SS + cl] = sv;
      }
    }
    __syncthreads();

    // online softmax per row (vectorized row scan)
    if (tid < 64) {
      int row = tid;
      const v4f* srow = (const v4f*)(Sls + row * SS);
      float mold = mrow[row], mnew = mold;
#pragma unroll
      for (int j4 = 0; j4 < 16; ++j4) {
        v4f x = srow[j4];
        mnew = fmaxf(mnew, fmaxf(fmaxf(x[0], x[1]), fmaxf(x[2], x[3])));
      }
      float corr = __expf(mold - mnew);
      float sum = 0.0f;
#pragma unroll
      for (int j8 = 0; j8 < 8; ++j8) {
        v4f x0 = srow[j8 * 2], x1 = srow[j8 * 2 + 1];
        v8h p;
#pragma unroll
        for (int e = 0; e < 4; ++e) {
          float p0 = __expf(x0[e] - mnew), p1 = __expf(x1[e] - mnew);
          sum += p0 + p1;
          p[e] = (_Float16)p0; p[4 + e] = (_Float16)p1;
        }
        *(v8h*)(Pl + row * ST + j8 * 8) = p;
      }
      lrow[row] = lrow[row] * corr + sum;
      mrow[row] = mnew;
      crow[row] = corr;
    }
    __syncthreads();

    // O = O*corr + P @ V
#pragma unroll
    for (int nt = 0; nt < 4; ++nt)
#pragma unroll
      for (int r = 0; r < 8; ++r) Oa[nt][r] *= crow[w * 16 + r + hi8];
#pragma unroll
    for (int nt = 0; nt < 4; ++nt)
#pragma unroll
      for (int k0 = 0; k0 < 64; k0 += 32) {
        v16h a = load_a_frag(Pl, ST, arow, k0, lane);
        v16h b = load_b_frag(Vc, ST, nt * 16 + (lane & 15), k0, lane);
        Oa[nt] = __builtin_amdgcn_wmma_f32_16x16x32_f16(false, a, false, b, (short)0, Oa[nt], false, false);
      }
    __syncthreads();
    kPtr += kvStep;
    vPtr += kvStep;
  }

  float* yg = y + ((size_t)h * Lq + qb * 64) * dh;
#pragma unroll
  for (int nt = 0; nt < 4; ++nt)
#pragma unroll
    for (int r = 0; r < 8; ++r) {
      int rl = w * 16 + r + hi8;
      int cl = nt * 16 + (lane & 15);
      yg[(size_t)rl * dh + cl] = Oa[nt][r] / lrow[rl];
    }
}

// ---------------- STP linear recurrence: one WG per head ----------------
__global__ __launch_bounds__(256)
void stp_scan(const float* __restrict__ q, const float* __restrict__ k,
              const float* __restrict__ v, const float* __restrict__ Wst,
              const float* __restrict__ Lam, const float* __restrict__ Gam,
              float* __restrict__ y) {
  __shared__ float qv[64], kv[64], vv[64];
  __shared__ float part[64 * 4];
  int tid = threadIdx.x, h = blockIdx.x;
  int i = tid >> 2, jg = tid & 3, j0 = jg * 16;
  size_t pbase = ((size_t)h * 64 + i) * 64 + j0;
  float st[16], ret[16], gm[16], wsr[16];
#pragma unroll
  for (int jj = 0; jj < 16; ++jj) {
    st[jj]  = 0.0f;
    ret[jj] = 1.0f / (1.0f + __expf(Lam[pbase + jj]));  // 1 - sigmoid(x)
    gm[jj]  = Gam[pbase + jj];
    wsr[jj] = Wst[pbase + jj];
  }
  const float* qg = q + (size_t)h * Lq * dh;
  const float* kg = k + (size_t)h * Lq * dh;
  const float* vg = v + (size_t)h * Lq * dh;

  for (int t = 0; t < Lq; ++t) {
    if (tid < 64) {
      qv[tid] = qg[(size_t)t * dh + tid];
      kv[tid] = kg[(size_t)t * dh + tid] * SCALE;
      vv[tid] = vg[(size_t)t * dh + tid];
    }
    __syncthreads();
    float vi = vv[i];
    float p = 0.0f;
#pragma unroll
    for (int jj = 0; jj < 16; ++jj) {
      st[jj] = ret[jj] * st[jj] + gm[jj] * (vi * kv[j0 + jj]);
      p += (wsr[jj] + st[jj]) * qv[j0 + jj];
    }
    part[i * 4 + jg] = p;
    __syncthreads();
    if (tid < 64)
      y[(((size_t)h * Lq + t)) * dh + tid] =
          part[tid * 4] + part[tid * 4 + 1] + part[tid * 4 + 2] + part[tid * 4 + 3];
  }
}

// ---------------- gate layer 2: alpha = sigmoid(g @ w2 + b2), (L,16) ----------------
__global__ __launch_bounds__(256)
void gate2_alpha(const float* __restrict__ g, const float* __restrict__ w2,
                 const float* __restrict__ b2, float* __restrict__ alpha) {
  int l = blockIdx.x * 16 + (threadIdx.x >> 4);
  int h = threadIdx.x & 15;
  const v4f* gr = (const v4f*)(g + (size_t)l * (Dm / 2));
  float s = 0.0f;
  for (int k4 = 0; k4 < (Dm / 2) / 4; ++k4) {
    v4f x = gr[k4];
#pragma unroll
    for (int e = 0; e < 4; ++e) s += x[e] * w2[(size_t)(k4 * 4 + e) * Hh + h];
  }
  s += b2[h];
  alpha[(size_t)l * Hh + h] = 1.0f / (1.0f + __expf(-s));
}

// ---------------- gate blend -> (L,D) ----------------
__global__ void combine_gate(const float* __restrict__ alpha, const float* __restrict__ ysm,
                             const float* __restrict__ ystp, float* __restrict__ yc) {
  int idx = blockIdx.x * blockDim.x + threadIdx.x;
  if (idx >= Lq * Dm) return;
  int l = idx >> 10;
  int c = idx & 1023;
  int h = c >> 6, i = c & 63;
  float a = alpha[l * Hh + h];
  size_t src = ((size_t)h * Lq + l) * dh + i;
  yc[idx] = a * ysm[src] + (1.0f - a) * ystp[src];
}

extern "C" void kernel_launch(void* const* d_in, const int* in_sizes, int n_in,
                              void* d_out, int out_size, void* d_ws, size_t ws_size,
                              hipStream_t stream) {
  const float* x      = (const float*)d_in[0];
  const float* Wqkv_w = (const float*)d_in[1];
  const float* Wqkv_b = (const float*)d_in[2];
  const float* out_w  = (const float*)d_in[3];
  const float* out_b  = (const float*)d_in[4];
  const float* W_stat = (const float*)d_in[5];
  const float* Lambda = (const float*)d_in[6];
  const float* Gamma  = (const float*)d_in[7];
  const float* gw1    = (const float*)d_in[8];
  const float* gb1    = (const float*)d_in[9];
  const float* gw2    = (const float*)d_in[10];
  const float* gb2    = (const float*)d_in[11];
  (void)in_sizes; (void)n_in; (void)out_size; (void)ws_size;

  float* ws = (float*)d_ws;
  size_t off = 0;
  float* qkv   = ws + off; off += (size_t)Lq * 3 * Dm;
  float* q     = ws + off; off += (size_t)Lq * Dm;
  float* k     = ws + off; off += (size_t)Lq * Dm;
  float* v     = ws + off; off += (size_t)Lq * Dm;
  float* ysm   = ws + off; off += (size_t)Lq * Dm;
  float* ystp  = ws + off; off += (size_t)Lq * Dm;
  float* g     = ws + off; off += (size_t)Lq * (Dm / 2);
  float* alpha = ws + off; off += (size_t)Lq * Hh;
  float* yc    = ws + off; off += (size_t)Lq * Dm;
  float* out   = (float*)d_out;

  gemm_wmma<<<dim3(Lq / 64, 3 * Dm / 128), 256, 0, stream>>>(x, Wqkv_w, Wqkv_b, qkv,
                                                             Lq, 3 * Dm, Dm, 0);
  repack_qkv<<<(Lq * 3 * Dm + 255) / 256, 256, 0, stream>>>(qkv, q, k, v);
  flash_attn<<<dim3(Lq / 64, Hh), 128, 0, stream>>>(q, k, v, ysm);
  stp_scan<<<Hh, 256, 0, stream>>>(q, k, v, W_stat, Lambda, Gamma, ystp);
  gemm_wmma<<<dim3(Lq / 64, (Dm / 2) / 128), 256, 0, stream>>>(x, gw1, gb1, g,
                                                               Lq, Dm / 2, Dm, 1);
  gate2_alpha<<<Lq / 16, 256, 0, stream>>>(g, gw2, gb2, alpha);
  combine_gate<<<(Lq * Dm + 255) / 256, 256, 0, stream>>>(alpha, ysm, ystp, yc);
  gemm_wmma<<<dim3(Lq / 64, Dm / 128), 256, 0, stream>>>(yc, out_w, out_b, out,
                                                         Lq, Dm, Dm, 0);
}
